// HyperGMULayer_88837103550771
// MI455X (gfx1250) — matrix operand order repaired and checked by reading, not compiled, not run on hardware
//
#include <hip/hip_runtime.h>
#include <hip/hip_bf16.h>
#include <math.h>

typedef __attribute__((ext_vector_type(16))) __bf16        v16bf;
typedef __attribute__((ext_vector_type(8)))  float         v8f;
typedef __attribute__((ext_vector_type(4)))  float         v4f;
typedef __attribute__((ext_vector_type(4)))  unsigned int  v4u;

#define NB    64      // batch
#define NOC   128     // out channels
#define NP    25      // patch features
#define NL    784     // locations (28*28)
#define NOUT  6400    // hyper output dim
#define COLS  16      // (b,l) columns per block
#define EPSN  1e-4f
#define DECAY 1e-4f

// round-to-nearest-even float -> bf16 bits
static __device__ __forceinline__ unsigned short f2bf(float x) {
    union { float f; unsigned u; } c; c.f = x;
    unsigned r = c.u + 0x7fffu + ((c.u >> 16) & 1u);
    return (unsigned short)(r >> 16);
}

union AV { v16bf v; v4u q[2]; unsigned short u[16]; };

// ---------------- kernel 1: g[j] = mean_b y2[b, j] ----------------
__global__ __launch_bounds__(256) void mean_kernel(const float* __restrict__ y2,
                                                   float* __restrict__ g) {
    int j = blockIdx.x * blockDim.x + threadIdx.x;
    if (j < NL) {
        float s = 0.f;
        for (int b = 0; b < NB; ++b) s += y2[b * NL + j];
        g[j] = s * (1.0f / NB);
    }
}

// ------- kernel 2: wvec[o] = dot(g, hyper_w[o,:]) + hyper_b[o] -------
__global__ __launch_bounds__(256) void hyper_kernel(const float* __restrict__ g,
                                                    const float* __restrict__ hw,
                                                    const float* __restrict__ hb,
                                                    float* __restrict__ wvec) {
    int wave = threadIdx.x >> 5;
    int lane = threadIdx.x & 31;
    int o = blockIdx.x * 8 + wave;               // 800 blocks * 8 waves = 6400
    if (o >= NOUT) return;
    const float* row = hw + (size_t)o * NL;
    float partial = 0.f;
    for (int j = lane; j < NL; j += 32) partial += g[j] * row[j];
    for (int off = 16; off > 0; off >>= 1)
        partial += __shfl_down(partial, off, 32);
    if (lane == 0) wvec[o] = partial + hb[o];
}

// ------- kernel 3: per-oc 3x3 algebra + bf16 A-matrix staging -------
// R = (2*A - A*G*A)/NP with G = X^T X, A = (G + DECAY*I)^-1.
// Store 6 symmetric coeffs (off-diagonals doubled), padded to 8 floats/oc:
//   q = c0*u0^2 + c1*u1^2 + c2*u2^2 + c3*u0*u1 + c4*u0*u2 + c5*u1*u2
// Wbf[256][32]: row r = 2*oc+s holds weights X[:,1+s] (k<25), zero-pad to 32.
__global__ __launch_bounds__(128) void prep_kernel(const float* __restrict__ wvec,
                                                   float* __restrict__ Bq,
                                                   unsigned short* __restrict__ Wbf) {
    int oc = threadIdx.x;
    if (oc >= NOC) return;
    float X1[NP], X2[NP];
    for (int p = 0; p < NP; ++p) {
        X1[p] = wvec[(oc * NP + p) * 2 + 0];
        X2[p] = wvec[(oc * NP + p) * 2 + 1];
    }
    float G[3][3];
    float s1 = 0, s2 = 0, s11 = 0, s22 = 0, s12 = 0;
    for (int p = 0; p < NP; ++p) {
        s1 += X1[p]; s2 += X2[p];
        s11 += X1[p] * X1[p]; s22 += X2[p] * X2[p]; s12 += X1[p] * X2[p];
    }
    G[0][0] = (float)NP; G[0][1] = s1;  G[0][2] = s2;
    G[1][0] = s1;        G[1][1] = s11; G[1][2] = s12;
    G[2][0] = s2;        G[2][1] = s12; G[2][2] = s22;
    // A = inv(G + DECAY*I) via adjugate
    float M00 = G[0][0] + DECAY, M01 = G[0][1], M02 = G[0][2];
    float M10 = G[1][0], M11 = G[1][1] + DECAY, M12 = G[1][2];
    float M20 = G[2][0], M21 = G[2][1], M22 = G[2][2] + DECAY;
    float c00 =  (M11 * M22 - M12 * M21);
    float c01 = -(M10 * M22 - M12 * M20);
    float c02 =  (M10 * M21 - M11 * M20);
    float det = M00 * c00 + M01 * c01 + M02 * c02;
    float id  = 1.0f / det;
    float A[3][3];
    A[0][0] = c00 * id;
    A[1][0] = c01 * id;
    A[2][0] = c02 * id;
    A[0][1] = -(M01 * M22 - M02 * M21) * id;
    A[1][1] =  (M00 * M22 - M02 * M20) * id;
    A[2][1] = -(M00 * M21 - M01 * M20) * id;
    A[0][2] =  (M01 * M12 - M02 * M11) * id;
    A[1][2] = -(M00 * M12 - M02 * M10) * id;
    A[2][2] =  (M00 * M11 - M01 * M10) * id;
    // T = A*G ; R = (2A - T*A)/NP
    float T[3][3], R[3][3];
    for (int i = 0; i < 3; ++i)
        for (int j = 0; j < 3; ++j)
            T[i][j] = A[i][0] * G[0][j] + A[i][1] * G[1][j] + A[i][2] * G[2][j];
    for (int i = 0; i < 3; ++i)
        for (int j = 0; j < 3; ++j) {
            float ta = T[i][0] * A[0][j] + T[i][1] * A[1][j] + T[i][2] * A[2][j];
            R[i][j] = (2.0f * A[i][j] - ta) * (1.0f / NP);
        }
    float* dst = Bq + oc * 8;
    dst[0] = R[0][0];
    dst[1] = R[1][1];
    dst[2] = R[2][2];
    dst[3] = R[0][1] + R[1][0];
    dst[4] = R[0][2] + R[2][0];
    dst[5] = R[1][2] + R[2][1];
    dst[6] = 0.f; dst[7] = 0.f;
    // bf16 A-operand rows (K padded 25 -> 32 with zeros)
    for (int k = 0; k < 32; ++k) {
        Wbf[(2 * oc + 0) * 32 + k] = (k < NP) ? f2bf(X1[k]) : (unsigned short)0;
        Wbf[(2 * oc + 1) * 32 + k] = (k < NP) ? f2bf(X2[k]) : (unsigned short)0;
    }
}

// ---------------- kernel 4: fused unfold/normalize/WMMA/activation ----------------
__global__ __launch_bounds__(256) void main_kernel(const float* __restrict__ y2,
                                                   const float* __restrict__ noise,
                                                   const unsigned short* __restrict__ Wbf,
                                                   const float* __restrict__ Bq,
                                                   float* __restrict__ out) {
    __shared__ float yraw[32][COLS];
    __shared__ __align__(16) unsigned short ybf[COLS][32];   // column-major: 64B/column
    __shared__ float ysum[COLS], ype[COLS], yinv[COLS];

    const int tid  = threadIdx.x;
    const int col0 = blockIdx.x * COLS;      // 784 % 16 == 0 -> whole block shares b
    const int b    = col0 / NL;
    const int l0   = col0 % NL;

    // Phase 1a: raw patch values (unfold with PAD=2) + eps*noise -> LDS
    for (int idx = tid; idx < 32 * COLS; idx += 256) {
        int j = idx >> 5;                    // column 0..15
        int k = idx & 31;                    // patch feature 0..31
        float v = 0.f;
        if (k < NP) {
            int l  = l0 + j;
            int py = l / 28, px = l % 28;
            int sy = py + (k / 5) - 2;
            int sx = px + (k % 5) - 2;
            float img = 0.f;
            if (sy >= 0 && sy < 28 && sx >= 0 && sx < 28)
                img = y2[b * NL + sy * 28 + sx];
            v = img + EPSN * noise[((size_t)b * NP + k) * NL + l];
        }
        yraw[k][j] = v;
    }
    __syncthreads();

    // Phase 1b: per-column ddof=1 std; u0 and exp(-||ynorm||^2/NP)
    if (tid < COLS) {
        float s = 0.f, ss = 0.f;
        for (int k = 0; k < NP; ++k) { float v = yraw[k][tid]; s += v; ss += v * v; }
        float var = (ss - s * s * (1.0f / NP)) * (1.0f / (NP - 1));
        float inv = rsqrtf(var);
        yinv[tid] = inv;
        ysum[tid] = s * inv;
        ype[tid]  = expf(-ss * inv * inv * (1.0f / NP));
    }
    __syncthreads();

    // Phase 1c: normalize + bf16, column-major so each half-K slice is contiguous
    for (int idx = tid; idx < 32 * COLS; idx += 256) {
        int j = idx >> 5, k = idx & 31;
        ybf[j][k] = f2bf(yraw[k][j] * yinv[j]);
    }
    __syncthreads();

    // Phase 2: WMMA. wave w covers row tiles 2w, 2w+1 (rows = 2*oc+s).
    const int wave = tid >> 5;
    const int lane = tid & 31;
    const int nloc = lane & 15;              // column within tile
    const int hi   = lane >> 4;              // lane half

    // B operand: lane half hi holds K = 8*hi + e (e<8), 16 + 8*hi + (e-8) (e>=8) for column nloc (2x ds_load_b128)
    AV bu;
    bu.q[0] = *(const v4u*)&ybf[nloc][hi * 8];
    bu.q[1] = *(const v4u*)&ybf[nloc][16 + hi * 8];
    const v16bf bvec = bu.v;

    const float u0 = ysum[nloc];
    const float pe = ype[nloc];
    const int   l  = l0 + nloc;
    const float kE1  = 0.36787944117144233f;          // e^-1
    const float kScl = 1.0f / (1.0f - kE1);
    const v4u* __restrict__ WbfQ = (const v4u*)Wbf;   // 8 shorts per v4u
    float* __restrict__ outB = out + (size_t)b * NOC * NL + l;

    #pragma unroll
    for (int t = 0; t < 2; ++t) {
        const int tile = wave * 2 + t;                // 0..15
        const int row  = tile * 16 + nloc;            // A-matrix row (M = lane&15)
        // A operand (16-bit A layout): lanes<16 K in {0..7,16..23}, lanes>=16 {8..15,24..31}
        AV au;
        au.q[0] = WbfQ[row * 4 + hi];                 // K = klo..klo+7
        au.q[1] = WbfQ[row * 4 + 2 + hi];             // K = khi..khi+7

        v8f c = {};
        c = __builtin_amdgcn_wmma_f32_16x16x32_bf16(
                /*neg_a=*/false, au.v, /*neg_b=*/false, bvec,
                /*c_mod=*/(short)0, c, /*reuse_a=*/false, /*reuse_b=*/false);

        // D layout: VGPR r <-> M = 8*hi + r; row pair (2i,2i+1) = (s=0,s=1) of oc
        #pragma unroll
        for (int i = 0; i < 4; ++i) {
            const int oc = tile * 8 + hi * 4 + i;
            const float u1 = c[2 * i];
            const float u2 = c[2 * i + 1];
            const v4f ca = *(const v4f*)(Bq + oc * 8);
            const v4f cb = *(const v4f*)(Bq + oc * 8 + 4);
            float q = ca.x * u0 * u0 + ca.y * u1 * u1 + ca.z * u2 * u2
                    + ca.w * u0 * u1 + cb.x * u0 * u2 + cb.y * u1 * u2;
            outB[(size_t)oc * NL] = (pe * expf(q) - kE1) * kScl - 0.5f;
        }
    }
}

extern "C" void kernel_launch(void* const* d_in, const int* in_sizes, int n_in,
                              void* d_out, int out_size, void* d_ws, size_t ws_size,
                              hipStream_t stream) {
    const float* y2      = (const float*)d_in[0];   // [64,1,28,28]
    const float* noise   = (const float*)d_in[1];   // [64,25,784]
    const float* hyper_w = (const float*)d_in[2];   // [6400,784]
    const float* hyper_b = (const float*)d_in[3];   // [6400]
    float* out = (float*)d_out;                     // [64,128,28,28]

    char* ws = (char*)d_ws;
    float*          g    = (float*)(ws + 0);              // 784 f
    float*          wvec = (float*)(ws + 4096);           // 6400 f
    float*          Bq   = (float*)(ws + 32768);          // 128*8 f
    unsigned short* Wbf  = (unsigned short*)(ws + 40960); // 256*32 bf16 bits

    mean_kernel <<<(NL + 255) / 256, 256, 0, stream>>>(y2, g);
    hyper_kernel<<<NOUT / 8, 256, 0, stream>>>(g, hyper_w, hyper_b, wvec);
    prep_kernel <<<1, 128, 0, stream>>>(wvec, Bq, Wbf);
    main_kernel <<<(NB * NL) / COLS, 256, 0, stream>>>(y2, noise, Wbf, Bq, out);
}
